// WOP_14602888806627
// MI455X (gfx1250) — matrix operand order repaired and checked by reading, not compiled, not run on hardware
//
#include <hip/hip_runtime.h>

// ---------------- types for WMMA ----------------
typedef __attribute__((ext_vector_type(16))) __bf16          v16bf;
typedef __attribute__((ext_vector_type(16))) unsigned short  v16u;
typedef __attribute__((ext_vector_type(8)))  unsigned short  v8u;
typedef __attribute__((ext_vector_type(8)))  float           v8f;

// fp32 -> bf16 (round to nearest even)
__device__ __forceinline__ unsigned short f2bf(float f) {
    unsigned int u = __float_as_uint(f);
    unsigned int r = u + 0x7FFFu + ((u >> 16) & 1u);
    return (unsigned short)(r >> 16);
}

__device__ __forceinline__ float sigmoidf_(float x) {
    return 1.0f / (1.0f + __expf(-x));
}

// assemble a 16-element bf16 A-fragment from two contiguous 8-element runs
__device__ __forceinline__ v16bf mkfrag(v8u lo, v8u hi) {
    v16u r;
    #pragma unroll
    for (int i = 0; i < 8; ++i) { r[i] = lo[i]; r[8 + i] = hi[i]; }
    return __builtin_bit_cast(v16bf, r);
}

// ============================================================
// fp32 -> bf16 conversion with zero padding: src (R,K) row-major
// -> dst (Rp,Kp) row-major, zeros outside.
// ============================================================
__global__ __launch_bounds__(256) void convert_pad_bf16(
    const float* __restrict__ src, unsigned short* __restrict__ dst,
    int R, int K, int Kp, long total)
{
    long idx = (long)blockIdx.x * 256 + threadIdx.x;
    if (idx >= total) return;
    int r = (int)(idx / Kp);
    int c = (int)(idx % Kp);
    float v = (r < R && c < K) ? src[(long)r * K + c] : 0.0f;
    dst[idx] = f2bf(v);
}

// ============================================================
// GEMM: C[M,N] = A[M,K] @ W[N,K]^T (+bias[N]) (+tanh)
// A,W are pre-converted, zero-padded bf16 (Mp x Kp), (Np x Kp).
// Each wave computes a 32x32 C tile: 4 WMMAs per 32-wide K step,
// fragments loaded straight from global (b128s), no LDS, no
// conditionals in the K loop. Wave-uniform early exit keeps EXEC
// all-ones for WMMA.
// ============================================================
__global__ __launch_bounds__(256) void gemm_bf16_wmma(
    const unsigned short* __restrict__ A, const unsigned short* __restrict__ W,
    const float* __restrict__ bias, float* __restrict__ C,
    int M, int N, int Kp, int ldc, int act, int tn, int ntiles)
{
    const int wave = threadIdx.x >> 5;
    const int lane = threadIdx.x & 31;
    const int gw = blockIdx.x * 8 + wave;
    if (gw >= ntiles) return;                 // whole-wave exit

    const int bm = gw / tn;
    const int bn = gw % tn;
    const int m0 = bm * 32;
    const int n0 = bn * 32;
    const int half = lane >> 4;               // 0/1
    const int l16  = lane & 15;

    // A frag: lane row m, elements K in {half*8..+7} and {16+half*8..+7}
    const unsigned short* pa0 = A + (long)(m0 + l16) * Kp + half * 8;
    const unsigned short* pa1 = pa0 + (long)16 * Kp;
    // B frag: lane row n (column of C), elements K in {half*8..+7} and {16+half*8..+7}
    const unsigned short* pb0 = W + (long)(n0 + l16) * Kp + half * 8;
    const unsigned short* pb1 = pb0 + (long)16 * Kp;

    v8f acc00 = {}, acc01 = {}, acc10 = {}, acc11 = {};

    for (int k = 0; k < Kp; k += 32) {
        v16bf a0 = mkfrag(*(const v8u*)(pa0 + k), *(const v8u*)(pa0 + k + 16));
        v16bf a1 = mkfrag(*(const v8u*)(pa1 + k), *(const v8u*)(pa1 + k + 16));
        v16bf b0 = mkfrag(*(const v8u*)(pb0 + k), *(const v8u*)(pb0 + k + 16));
        v16bf b1 = mkfrag(*(const v8u*)(pb1 + k), *(const v8u*)(pb1 + k + 16));
        acc00 = __builtin_amdgcn_wmma_f32_16x16x32_bf16(false, a0, false, b0, (short)0, acc00, false, false);
        acc01 = __builtin_amdgcn_wmma_f32_16x16x32_bf16(false, a0, false, b1, (short)0, acc01, false, false);
        acc10 = __builtin_amdgcn_wmma_f32_16x16x32_bf16(false, a1, false, b0, (short)0, acc10, false, false);
        acc11 = __builtin_amdgcn_wmma_f32_16x16x32_bf16(false, a1, false, b1, (short)0, acc11, false, false);
    }

    // epilogue: C/D layout: vgpr r -> M = r + half*8, N = l16
    #pragma unroll
    for (int mi = 0; mi < 2; ++mi) {
        #pragma unroll
        for (int ni = 0; ni < 2; ++ni) {
            const v8f acc = (mi == 0) ? (ni == 0 ? acc00 : acc01)
                                      : (ni == 0 ? acc10 : acc11);
            const int gn = n0 + ni * 16 + l16;
            if (gn >= N) continue;
            const float bv = bias ? bias[gn] : 0.0f;
            #pragma unroll
            for (int r = 0; r < 8; ++r) {
                const int gm = m0 + mi * 16 + r + half * 8;
                if (gm < M) {
                    float v = acc[r] + bv;
                    if (act) v = tanhf(v);
                    C[(long)gm * ldc + gn] = v;
                }
            }
        }
    }
}

// ============================================================
// LSTM recurrence for one direction/layer. One block per sequence.
// G (B,T,200) holds precomputed x@Wih.T. Output masked h into
// out[(b*T+t)*outStride + j] (pointer pre-offset for direction).
// ============================================================
#define H2C 50
#define G4C 200

__global__ __launch_bounds__(256) void lstm_dir_kernel(
    const float* __restrict__ G, const float* __restrict__ Whh,
    const float* __restrict__ bih, const float* __restrict__ bhh,
    const int* __restrict__ lens, float* __restrict__ out,
    int T, int outStride, int reverse)
{
    __shared__ float sW[G4C * H2C];   // 40 KB
    __shared__ float sBias[G4C];
    __shared__ float sH[H2C];
    __shared__ float sC[H2C];
    __shared__ float sG[G4C];

    const int b = blockIdx.x;
    const int tid = threadIdx.x;

    for (int i = tid; i < G4C * H2C; i += 256) sW[i] = Whh[i];
    if (tid < G4C) sBias[tid] = bih[tid] + bhh[tid];
    if (tid < H2C) { sH[tid] = 0.0f; sC[tid] = 0.0f; }
    __syncthreads();

    const int len = lens[b];
    for (int s = 0; s < T; ++s) {
        const int t = reverse ? (T - 1 - s) : s;
        if (tid < G4C) {
            float a = G[((long)b * T + t) * G4C + tid] + sBias[tid];
            const float* wr = &sW[tid * H2C];
            #pragma unroll 10
            for (int k = 0; k < H2C; ++k) a += wr[k] * sH[k];
            sG[tid] = a;
        }
        __syncthreads();
        if (tid < H2C) {
            const float ig = sigmoidf_(sG[tid]);
            const float fg = sigmoidf_(sG[H2C + tid]);
            const float gg = tanhf(sG[2 * H2C + tid]);
            const float og = sigmoidf_(sG[3 * H2C + tid]);
            const float cn = fg * sC[tid] + ig * gg;
            const float hn = og * tanhf(cn);
            const bool valid = (t < len);
            out[((long)b * T + t) * outStride + tid] = valid ? hn : 0.0f;
            if (valid) { sC[tid] = cn; sH[tid] = hn; }
        }
        __syncthreads();
    }
}

// wenc_hpu[n,:] = hpu_out[n, l_hpu[n]-1, :]
__global__ void gather_last_kernel(const float* __restrict__ hpu_out,
                                   const int* __restrict__ l_hpu,
                                   float* __restrict__ wenc_hpu, int T)
{
    const int n = blockIdx.x, tid = threadIdx.x;
    if (tid < 100) {
        const int t = l_hpu[n] - 1;
        wenc_hpu[(long)n * 100 + tid] = hpu_out[((long)n * T + t) * 100 + tid];
    }
}

// exclusive cumsum of l_hs (B=256, trivial)
__global__ void offsets_kernel(const int* __restrict__ l_hs,
                               int* __restrict__ offs, int B)
{
    if (blockIdx.x == 0 && threadIdx.x == 0) {
        int acc = 0;
        for (int b = 0; b < B; ++b) { offs[b] = acc; acc += l_hs[b]; }
    }
}

// ob[b,w,:] = (idx<l_hs[b]) ? wenc_hpu[clip(off[b]+idx)] : 0
__global__ void build_ob_kernel(const float* __restrict__ wenc_hpu,
                                const int* __restrict__ offs,
                                const int* __restrict__ l_hs,
                                const int* __restrict__ wn,
                                const int* __restrict__ wc,
                                float* __restrict__ ob, int Nhpu)
{
    const int b = blockIdx.x, tid = threadIdx.x;
    if (tid >= 100) return;
    const int lb = l_hs[b], off = offs[b], w_n = wn[b];
    for (int w = 0; w < 20; ++w) {
        const int idx = (w < w_n) ? wc[b * 20 + w] : 0;
        float v = 0.0f;
        if (idx < lb) {
            int row = off + idx;
            row = min(max(row, 0), Nhpu - 1);
            v = wenc_hpu[(long)row * 100 + tid];
        }
        ob[((long)b * 20 + w) * 100 + tid] = v;
    }
}

// attention + softmax + context; one block (128 thr) per (b, j)
__global__ __launch_bounds__(128) void attn_kernel(
    const float* __restrict__ q, const float* __restrict__ wenc_n,
    const float* __restrict__ ob, const int* __restrict__ l_n,
    float* __restrict__ c_n, int T)
{
    const int b = blockIdx.x / 20;
    const int j = blockIdx.x % 20;
    const int tid = threadIdx.x;
    __shared__ float sOb[100];
    __shared__ float sP[128];
    __shared__ float red[128];

    if (tid < 100) sOb[tid] = ob[((long)b * 20 + j) * 100 + tid];
    __syncthreads();

    const int len = l_n[b];
    float att = -1.0e10f;                 // NEG mask
    if (tid < len) {
        float acc = 0.0f;
        const float* qr = &q[((long)b * T + tid) * 100];
        #pragma unroll 4
        for (int h = 0; h < 100; ++h) acc += qr[h] * sOb[h];
        att = acc;
    }
    red[tid] = att;
    __syncthreads();
    for (int s = 64; s > 0; s >>= 1) {
        if (tid < s) red[tid] = fmaxf(red[tid], red[tid + s]);
        __syncthreads();
    }
    const float mx = red[0];
    __syncthreads();
    const float e = __expf(att - mx);
    red[tid] = e;
    __syncthreads();
    for (int s = 64; s > 0; s >>= 1) {
        if (tid < s) red[tid] += red[tid + s];
        __syncthreads();
    }
    sP[tid] = e / red[0];
    __syncthreads();

    if (tid < 100) {
        float acc = 0.0f;
        for (int t = 0; t < T; ++t)
            acc += sP[t] * wenc_n[((long)b * T + t) * 100 + tid];
        c_n[((long)b * 20 + j) * 100 + tid] = acc;
    }
}

// ---------------- host-side orchestration ----------------
static inline int rup(int x, int a) { return ((x + a - 1) / a) * a; }

static void run_gemm(const float* A, const float* W, const float* bias, float* C,
                     int M, int N, int K, int ldc, int act,
                     unsigned short* Abf, unsigned short* Wbf, bool convA,
                     hipStream_t s)
{
    const int Mp = rup(M, 32), Np = rup(N, 32), Kp = rup(K, 32);
    if (convA) {
        const long tot = (long)Mp * Kp;
        convert_pad_bf16<<<dim3((unsigned)((tot + 255) / 256)), 256, 0, s>>>(A, Abf, M, K, Kp, tot);
    }
    {
        const long tot = (long)Np * Kp;
        convert_pad_bf16<<<dim3((unsigned)((tot + 255) / 256)), 256, 0, s>>>(W, Wbf, N, K, Kp, tot);
    }
    const int tn = Np / 32;
    const int ntiles = (Mp / 32) * tn;
    gemm_bf16_wmma<<<dim3((ntiles + 7) / 8), 256, 0, s>>>(Abf, Wbf, bias, C, M, N, Kp, ldc, act, tn, ntiles);
}

extern "C" void kernel_launch(void* const* d_in, const int* in_sizes, int n_in,
                              void* d_out, int out_size, void* d_ws, size_t ws_size,
                              hipStream_t stream)
{
    const int B = 256, T = 128, MLHPU = 12;
    (void)n_in; (void)out_size; (void)ws_size;

    const float* wemb_n   = (const float*)d_in[0];
    const float* wemb_hpu = (const float*)d_in[1];
    const float* n_Wih0   = (const float*)d_in[2];
    const float* n_Wih1   = (const float*)d_in[3];
    const float* n_Whh    = (const float*)d_in[4];
    const float* n_bih    = (const float*)d_in[5];
    const float* n_bhh    = (const float*)d_in[6];
    const float* h_Wih0   = (const float*)d_in[7];
    const float* h_Wih1   = (const float*)d_in[8];
    const float* h_Whh    = (const float*)d_in[9];
    const float* h_bih    = (const float*)d_in[10];
    const float* h_bhh    = (const float*)d_in[11];
    const float* W_att_w  = (const float*)d_in[12];
    const float* W_att_b  = (const float*)d_in[13];
    const float* W_c_w    = (const float*)d_in[14];
    const float* W_c_b    = (const float*)d_in[15];
    const float* W_hs_w   = (const float*)d_in[16];
    const float* W_hs_b   = (const float*)d_in[17];
    const float* wo0_w    = (const float*)d_in[18];
    const float* wo0_b    = (const float*)d_in[19];
    const float* wo1_w    = (const float*)d_in[20];
    const float* wo1_b    = (const float*)d_in[21];
    const int*   l_n      = (const int*)d_in[22];
    const int*   l_hpu    = (const int*)d_in[23];
    const int*   l_hs     = (const int*)d_in[24];
    const int*   wn       = (const int*)d_in[25];
    const int*   wc       = (const int*)d_in[26];

    const int  Nh  = in_sizes[23];        // number of hpu sequences
    const long MnT = (long)B * T;         // 32768
    const long MhT = (long)Nh * MLHPU;

    // ---- carve workspace ----
    size_t off = 0;
    char* base = (char*)d_ws;
    auto carve = [&](size_t bytes) -> void* {
        void* p = base + off;
        off += (bytes + 255) & ~(size_t)255;
        return p;
    };
    const long maxM  = (MnT > MhT) ? MnT : MhT;
    const long maxMp = rup((int)maxM, 32);
    float* Gf       = (float*)carve((size_t)maxM * 200 * 4);
    float* Gb       = (float*)carve((size_t)maxM * 200 * 4);
    float* l0n      = (float*)carve((size_t)MnT * 100 * 4);
    float* wenc_n   = (float*)carve((size_t)MnT * 100 * 4);
    float* l0h      = (float*)carve((size_t)MhT * 100 * 4);
    float* hpu_out  = (float*)carve((size_t)MhT * 100 * 4);
    float* wenc_hpu = (float*)carve((size_t)Nh * 100 * 4);
    int*   offs     = (int*)  carve((size_t)B * 4);
    float* ob       = (float*)carve((size_t)B * 20 * 100 * 4);
    float* qbuf     = (float*)carve((size_t)MnT * 100 * 4);
    float* c_n      = (float*)carve((size_t)B * 20 * 100 * 4);
    float* vec      = (float*)carve((size_t)B * 20 * 200 * 4);
    float* h2       = (float*)carve((size_t)B * 20 * 100 * 4);
    unsigned short* Abf = (unsigned short*)carve((size_t)maxMp * 320 * 2); // bf16 A, Kp<=320
    unsigned short* Wbf = (unsigned short*)carve((size_t)224 * 320 * 2);   // bf16 W, Np<=224

    auto whh_off  = [](int l, int d) { return (size_t)(l * 2 + d) * 200 * 50; };
    auto bias_off = [](int l, int d) { return (size_t)(l * 2 + d) * 200; };

    // ================= question encoder (B=256, T=128) =================
    run_gemm(wemb_n, n_Wih0 + 0,         nullptr, Gf, (int)MnT, 200, 300, 200, 0, Abf, Wbf, true,  stream);
    run_gemm(wemb_n, n_Wih0 + 200 * 300, nullptr, Gb, (int)MnT, 200, 300, 200, 0, Abf, Wbf, false, stream);
    lstm_dir_kernel<<<B, 256, 0, stream>>>(Gf, n_Whh + whh_off(0, 0), n_bih + bias_off(0, 0),
                                           n_bhh + bias_off(0, 0), l_n, l0n + 0, T, 100, 0);
    lstm_dir_kernel<<<B, 256, 0, stream>>>(Gb, n_Whh + whh_off(0, 1), n_bih + bias_off(0, 1),
                                           n_bhh + bias_off(0, 1), l_n, l0n + 50, T, 100, 1);
    run_gemm(l0n, n_Wih1 + 0,         nullptr, Gf, (int)MnT, 200, 100, 200, 0, Abf, Wbf, true,  stream);
    run_gemm(l0n, n_Wih1 + 200 * 100, nullptr, Gb, (int)MnT, 200, 100, 200, 0, Abf, Wbf, false, stream);
    lstm_dir_kernel<<<B, 256, 0, stream>>>(Gf, n_Whh + whh_off(1, 0), n_bih + bias_off(1, 0),
                                           n_bhh + bias_off(1, 0), l_n, wenc_n + 0, T, 100, 0);
    lstm_dir_kernel<<<B, 256, 0, stream>>>(Gb, n_Whh + whh_off(1, 1), n_bih + bias_off(1, 1),
                                           n_bhh + bias_off(1, 1), l_n, wenc_n + 50, T, 100, 1);

    // ================= header encoder (Nh seqs, T=12) =================
    run_gemm(wemb_hpu, h_Wih0 + 0,         nullptr, Gf, (int)MhT, 200, 300, 200, 0, Abf, Wbf, true,  stream);
    run_gemm(wemb_hpu, h_Wih0 + 200 * 300, nullptr, Gb, (int)MhT, 200, 300, 200, 0, Abf, Wbf, false, stream);
    lstm_dir_kernel<<<Nh, 256, 0, stream>>>(Gf, h_Whh + whh_off(0, 0), h_bih + bias_off(0, 0),
                                            h_bhh + bias_off(0, 0), l_hpu, l0h + 0, MLHPU, 100, 0);
    lstm_dir_kernel<<<Nh, 256, 0, stream>>>(Gb, h_Whh + whh_off(0, 1), h_bih + bias_off(0, 1),
                                            h_bhh + bias_off(0, 1), l_hpu, l0h + 50, MLHPU, 100, 1);
    run_gemm(l0h, h_Wih1 + 0,         nullptr, Gf, (int)MhT, 200, 100, 200, 0, Abf, Wbf, true,  stream);
    run_gemm(l0h, h_Wih1 + 200 * 100, nullptr, Gb, (int)MhT, 200, 100, 200, 0, Abf, Wbf, false, stream);
    lstm_dir_kernel<<<Nh, 256, 0, stream>>>(Gf, h_Whh + whh_off(1, 0), h_bih + bias_off(1, 0),
                                            h_bhh + bias_off(1, 0), l_hpu, hpu_out + 0, MLHPU, 100, 0);
    lstm_dir_kernel<<<Nh, 256, 0, stream>>>(Gb, h_Whh + whh_off(1, 1), h_bih + bias_off(1, 1),
                                            h_bhh + bias_off(1, 1), l_hpu, hpu_out + 50, MLHPU, 100, 1);

    // ================= gathers =================
    gather_last_kernel<<<Nh, 128, 0, stream>>>(hpu_out, l_hpu, wenc_hpu, MLHPU);
    offsets_kernel<<<1, 32, 0, stream>>>(l_hs, offs, B);
    build_ob_kernel<<<B, 128, 0, stream>>>(wenc_hpu, offs, l_hs, wn, wc, ob, Nh);

    // ================= attention =================
    run_gemm(wenc_n, W_att_w, W_att_b, qbuf, (int)MnT, 100, 100, 100, 0, Abf, Wbf, true, stream);
    attn_kernel<<<B * 20, 128, 0, stream>>>(qbuf, wenc_n, ob, l_n, c_n, T);

    // ================= output head =================
    const int Mh = B * 20;  // 5120
    run_gemm(c_n, W_c_w,  W_c_b,  vec + 0,   Mh, 100, 100, 200, 0, Abf, Wbf, true, stream);
    run_gemm(ob,  W_hs_w, W_hs_b, vec + 100, Mh, 100, 100, 200, 0, Abf, Wbf, true, stream);
    run_gemm(vec, wo0_w,  wo0_b,  h2,        Mh, 100, 200, 100, 1, Abf, Wbf, true, stream);
    run_gemm(h2,  wo1_w,  wo1_b,  (float*)d_out, Mh, 7, 100, 7, 0, Abf, Wbf, true, stream);
}
